// minGRU_33904471835471
// MI455X (gfx1250) — matrix operand order repaired and checked by reading, not compiled, not run on hardware
//
#include <hip/hip_runtime.h>
#include <hip/hip_fp16.h>

// minGRU fused kernel for MI455X (gfx1250, wave32).
//
//   hg = W @ x + b          -> v_wmma_f32_16x16x32_f16 (f16 in, f32 acc)
//   lc = -softplus(gate); lv = -softplus(-gate) + log_g(h)
//   log_h_t = logaddexp(lc_t + log_h_{t-1}, lv_t)   (scan over L)
//   out = exp(log_h)
//
// x is streamed via the Tensor Data Mover (tensor_load_to_lds, TENSORcnt,
// depth-2 pipeline over 3 LDS buffers, TDM row padding for conflict-free
// LDS gathers). WMMA C tile is [l x o]: each lane owns one channel and 8
// consecutive l values -> in-register scan, ds_bpermute lane-half chaining,
// global flag/carry chain across L-chunks. Elementwise log-space math uses
// hardware v_exp_f32/v_log_f32 (co-executes with the XDL WMMA pipe).

typedef __attribute__((ext_vector_type(16))) _Float16 v16h;
typedef __attribute__((ext_vector_type(8)))  float    v8f;
typedef __attribute__((ext_vector_type(4)))  unsigned v4u;
typedef __attribute__((ext_vector_type(8)))  int      v8i;
typedef __attribute__((ext_vector_type(4)))  int      v4i;

#define BB     8
#define DD     512
#define HH     512
#define LL     8192
#define TILES  32          // HH / 16 channel tiles
#define NCH    32          // chunks along L
#define CHUNK  256         // LL / NCH
#define NTILE  16          // 16-wide l-tiles per chunk
#define NBLK   8           // 64-row d sub-tiles per l-tile
#define NQ     (NTILE * NBLK)          // 128 TDM transfers per chunk
#define XROW   18          // LDS dwords per 16-dword x row (TDM pad = 2)
#define XBUF_DW (64 * XROW)            // 1152 dwords per x buffer
#define STR    261         // padded LDS row stride (floats), conflict-free
#define WAVE_DW (3 * XBUF_DW + 2 * 16 * STR)
#define NEGINF (-1e30f)

// log(1+e) with e in (0,1]: argument of v_log_f32 is in (1,2] -> accurate.
__device__ __forceinline__ float lax(float a, float b) {      // logaddexp
  float m = fmaxf(a, b);
  float d = fminf(a, b) - m;
  return m + __logf(1.0f + __expf(d));
}
// softplus(x) = max(x,0) + log(1 + exp(-|x|))   (branchless, stable)
__device__ __forceinline__ float softplus_(float x) {
  return fmaxf(x, 0.0f) + __logf(1.0f + __expf(-fabsf(x)));
}
__device__ __forceinline__ float log_g_(float x) {
  return (x >= 0.0f) ? __logf(1.0f + x) : x;
}

// Issue one TDM transfer: 64 rows (d) x 16 cols (l) of f32 from global x
// into an LDS buffer, 2 pad dwords after every 16-dword row (stride 18).
__device__ __forceinline__ void tdm_issue(const float* gsrc, const float* lbuf) {
  unsigned long long ga = (unsigned long long)(uintptr_t)gsrc;
  unsigned lo = (unsigned)(uintptr_t)lbuf;   // LDS byte offset = low 32 bits
  v4u g0;
  g0.x = 1u;                                  // count=1 (valid user descriptor)
  g0.y = lo;                                  // lds_addr
  g0.z = (unsigned)ga;                        // global_addr[31:0]
  g0.w = (unsigned)((ga >> 32) & 0x01FFFFFFull) | (2u << 30);  // addr[56:32]|type=2
  v8i g1;
  g1[0] = (2 << 16)      // data_size = 4B
        | (1 << 20)      // pad_enable
        | (3 << 22)      // pad_interval code 3 -> every 16 dwords
        | (1 << 25);     // pad_amount  code 1 -> 2 dwords
  g1[1] = (int)((LL & 0xFFFF) << 16);                          // tensor_dim0 lo
  g1[2] = (int)(((LL >> 16) & 0xFFFF) | ((DD & 0xFFFF) << 16)); // dim0 hi|dim1 lo
  g1[3] = (int)(((DD >> 16) & 0xFFFF) | (16 << 16));            // dim1 hi|tile_dim0
  g1[4] = 64;                                                   // tile_dim1 rows
  g1[5] = LL;                                                   // dim0_stride lo
  g1[6] = 0;
  g1[7] = 0;
  v4i z4 = {0, 0, 0, 0};
#if defined(__clang_major__) && __clang_major__ >= 23
  v8i z8 = {0, 0, 0, 0, 0, 0, 0, 0};
  __builtin_amdgcn_tensor_load_to_lds(g0, g1, z4, z4, z8, 0);
#else
  __builtin_amdgcn_tensor_load_to_lds(g0, g1, z4, z4, 0);
#endif
}

__global__ __launch_bounds__(64)
void mingru_fused(const float* __restrict__ x, const float* __restrict__ W,
                  const float* __restrict__ bias, float* __restrict__ out,
                  float* __restrict__ carr, int* __restrict__ flags)
{
  __shared__ alignas(32) _Float16 lds_w[32 * DD];   // 16 h rows + 16 gate rows
  extern __shared__ float dsm[];                    // per-wave xbufs + (s,a)

  const int tid  = threadIdx.x;
  const int wav  = tid >> 5;
  const int lane = tid & 31;
  const int n    = lane & 15;    // A row (l offset) AND C column (channel)
  const int half = lane >> 4;

  const int bid  = blockIdx.x;
  const int cg   = bid / (BB * TILES);   // chunk-major dispatch order
  const int bt   = bid % (BB * TILES);
  const int b    = bt / TILES;
  const int tile = bt % TILES;
  const int o0   = tile * 16;
  const int c    = cg * 2 + wav;         // this wave's chunk index, 0..31

  // ---- stage W tile (h rows 0..15, gate rows 16..31) into LDS as f16 ----
  {
    const int row = tid >> 1;
    const int seg = tid & 1;
    const int grow = (row < 16) ? (o0 + row) : (HH + o0 + (row - 16));
    const float4* wp = (const float4*)(W + (size_t)grow * DD + seg * 256);
    _Float16* dst = lds_w + row * DD + seg * 256;
    #pragma unroll 4
    for (int i = 0; i < 64; ++i) {
      float4 v = wp[i];
      dst[4*i+0] = (_Float16)v.x;
      dst[4*i+1] = (_Float16)v.y;
      dst[4*i+2] = (_Float16)v.z;
      dst[4*i+3] = (_Float16)v.w;
    }
  }
  __syncthreads();

  float* xbuf = dsm + wav * WAVE_DW;         // 3 TDM buffers
  float* ls_s = xbuf + 3 * XBUF_DW;
  float* ls_a = ls_s + 16 * STR;

  const float bh = bias[o0 + n];
  const float bg = bias[HH + o0 + n];

  // wave-uniform base of this chunk inside x[b]
  const float* pxu = x + (size_t)b * DD * LL + (size_t)c * CHUNK;

  float sA = NEGINF;   // chunk-local running log_h (replicated per channel)
  float aA = 0.0f;     // chunk-local running sum of log-coeffs

  // prime the TDM pipeline (depth 2)
  tdm_issue(pxu, xbuf);
  tdm_issue(pxu + (size_t)64 * LL, xbuf + XBUF_DW);

  for (int tl = 0; tl < NTILE; ++tl) {
    v8f acc_h = {};
    v8f acc_g = {};

    for (int blk = 0; blk < NBLK; ++blk) {
      const int q = tl * NBLK + blk;
      if (q + 2 < NQ) {
        const int qn = q + 2;
        tdm_issue(pxu + (size_t)((qn & 7) * 64) * LL + (qn >> 3) * 16,
                  xbuf + (qn % 3) * XBUF_DW);
        __builtin_amdgcn_s_wait_tensorcnt(2);   // transfer q complete
      } else if (q + 2 == NQ) {
        __builtin_amdgcn_s_wait_tensorcnt(1);
      } else {
        __builtin_amdgcn_s_wait_tensorcnt(0);
      }

      const float* xb = xbuf + (q % 3) * XBUF_DW;
      #pragma unroll
      for (int ks2 = 0; ks2 < 2; ++ks2) {      // two K=32 steps per sub-tile
        const int d0l = ks2 * 32;              // row inside LDS buffer
        const int d0  = blk * 64 + d0l;        // absolute d for B fragments
        // A fragment: lane m holds row m, K={0..15} / {16..31}
        v16h Af;
        #pragma unroll
        for (int j = 0; j < 16; ++j) {
          const int kj = j + half * 16;
          Af[j] = (_Float16)xb[(d0l + kj) * XROW + n];
        }
        const v16h* Bh = (const v16h*)(lds_w + n * DD + d0 + half * 16);
        const v16h* Bg = (const v16h*)(lds_w + (16 + n) * DD + d0 + half * 16);
        acc_h = __builtin_amdgcn_wmma_f32_16x16x32_f16(false, Af, false, *Bh,
                                                       (short)0, acc_h, false, false);
        acc_g = __builtin_amdgcn_wmma_f32_16x16x32_f16(false, Af, false, *Bg,
                                                       (short)0, acc_g, false, false);
      }
    }

    // ---- elementwise gate math + per-lane local scan over 8 l values ----
    float la[8], ls[8];
    float runa = 0.0f, runs = NEGINF;
    #pragma unroll
    for (int j = 0; j < 8; ++j) {
      float hv = acc_h[j] + bh;
      float gv = acc_g[j] + bg;
      float lc = -softplus_(gv);
      float lv = -softplus_(-gv) + log_g_(hv);
      runa += lc;
      runs = lax(lc + runs, lv);
      la[j] = runa;
      ls[j] = runs;
    }

    // ---- chain lane halves: low lanes own l0..l7, high lanes l8..l15 ----
    float s_end0 = lax(sA + la[7], ls[7]);
    float a_end0 = aA + la[7];
    float s_part = __shfl(s_end0, n, 32);        // high lanes fetch partner end
    float a_part = __shfl(a_end0, n, 32);
    float s_base = half ? s_part : sA;
    float a_base = half ? a_part : aA;

    const int lloc0 = tl * 16 + half * 8;
    #pragma unroll
    for (int j = 0; j < 8; ++j) {
      ls_s[n * STR + lloc0 + j] = lax(s_base + la[j], ls[j]);
      ls_a[n * STR + lloc0 + j] = a_base + la[j];
    }

    float s_end1 = lax(s_base + la[7], ls[7]);
    float a_end1 = a_base + la[7];
    sA = __shfl(s_end1, n + 16, 32);             // broadcast high-lane end
    aA = __shfl(a_end1, n + 16, 32);
  }

  // ---- inter-chunk carry chain (publish before fix-up) ----
  const int sidx = bt * NCH + c;
  float cin = NEGINF;
  if (c > 0) {
    const int prev = sidx - 1;
    while (__hip_atomic_load(&flags[prev], __ATOMIC_ACQUIRE,
                             __HIP_MEMORY_SCOPE_AGENT) == 0) {
      __builtin_amdgcn_s_sleep(1);
    }
    cin = carr[(size_t)prev * 16 + n];
  }
  float cout = lax(cin + aA, sA);
  if (lane < 16) carr[(size_t)sidx * 16 + n] = cout;
  __threadfence();
  if (lane == 0)
    __hip_atomic_store(&flags[sidx], 1, __ATOMIC_RELEASE,
                       __HIP_MEMORY_SCOPE_AGENT);

  // ---- elementwise fix-up with carry-in, coalesced 128B row writes ----
  float* orow = out + (size_t)b * HH * LL + (size_t)c * CHUNK;
  for (int ch = 0; ch < 16; ++ch) {
    float cc = __shfl(cin, ch, 32);
    float* op = orow + (size_t)(o0 + ch) * LL;
    #pragma unroll
    for (int t = 0; t < CHUNK / 32; ++t) {
      const int lloc = t * 32 + lane;
      float s = ls_s[ch * STR + lloc];
      float a = ls_a[ch * STR + lloc];
      op[lloc] = __expf(lax(cc + a, s));
    }
  }
}

extern "C" void kernel_launch(void* const* d_in, const int* in_sizes, int n_in,
                              void* d_out, int out_size, void* d_ws, size_t ws_size,
                              hipStream_t stream) {
  const float* x  = (const float*)d_in[0];
  const float* W  = (const float*)d_in[1];
  const float* bb = (const float*)d_in[2];
  float* out      = (float*)d_out;

  // workspace: flags (B*TILES*NCH ints) then carries (16 floats per slot)
  const size_t nslots = (size_t)BB * TILES * NCH;      // 8192
  int*   flags = (int*)d_ws;
  float* carr  = (float*)((char*)d_ws + nslots * sizeof(int));

  hipMemsetAsync(d_ws, 0, nslots * sizeof(int), stream);   // reset flags per launch

  dim3 grid((NCH / 2) * BB * TILES);   // 16 chunk-groups * 256 (b,tile) = 4096
  dim3 block(64);                      // 2 waves sharing one W tile
  const size_t dyn = 2 * WAVE_DW * sizeof(float);          // per-wave LDS

  hipLaunchKernelGGL(mingru_fused, grid, block, dyn, stream,
                     x, W, bb, out, carr, flags);
}
